// LayerNormLSTMCell_14980845928861
// MI455X (gfx1250) — compile-verified, matrix operand order checked by reading, not run on hardware
//
#include <hip/hip_runtime.h>
#include <hip/hip_bf16.h>
#include <cstdint>

#define BATCH   4096
#define INPUT   512
#define HIDDEN  1024
#define NGATE   4096   // 4*HIDDEN
#define KDIM    1536   // INPUT + HIDDEN

#define BM 128
#define BN 128
#define BK 32
#define LDS_STRIDE 40  // 32 bf16 + 8 pad -> 80B row stride (16B aligned)
#define NKT (KDIM / BK)

typedef __attribute__((ext_vector_type(16))) __bf16 v16bf;
typedef __attribute__((ext_vector_type(8)))  float  v8f;
typedef __attribute__((ext_vector_type(4)))  int    v4i;

typedef __attribute__((address_space(1))) v4i* global_v4i_ptr;
typedef __attribute__((address_space(3))) v4i* lds_v4i_ptr;

#if __has_builtin(__builtin_amdgcn_global_load_async_to_lds_b128) && \
    __has_builtin(__builtin_amdgcn_s_wait_asynccnt)
#define USE_ASYNC_LDS 1
#else
#define USE_ASYNC_LDS 0
#endif

union Frag32B {
    uint4 u[2];
    v16bf v;
};

static __device__ __forceinline__ unsigned short f32_to_bf16_rne(float f) {
    union { float f; unsigned int u; } cvt;
    cvt.f = f;
    unsigned int u = cvt.u;
    u += 0x7fffu + ((u >> 16) & 1u);
    return (unsigned short)(u >> 16);
}

// -------------------------------------------------------------------------
// Kernel 1: concat two fp32 row-major matrices [rows x w0] and [rows x w1]
// into one bf16 row-major matrix [rows x (w0+w1)].
// -------------------------------------------------------------------------
__global__ void pack_bf16_kernel(const float* __restrict__ src0, int w0,
                                 const float* __restrict__ src1, int w1,
                                 unsigned short* __restrict__ dst, int rows) {
    const int wtot   = w0 + w1;
    const int total4 = rows * wtot / 4;
    for (int i = blockIdx.x * blockDim.x + threadIdx.x; i < total4;
         i += gridDim.x * blockDim.x) {
        const int idx = i * 4;
        const int row = idx / wtot;
        const int col = idx - row * wtot;
        const float* s = (col < w0) ? (src0 + (size_t)row * w0 + col)
                                    : (src1 + (size_t)row * w1 + (col - w0));
        float4 f = *(const float4*)s;
        ushort4 o;
        o.x = f32_to_bf16_rne(f.x);
        o.y = f32_to_bf16_rne(f.y);
        o.z = f32_to_bf16_rne(f.z);
        o.w = f32_to_bf16_rne(f.w);
        *(ushort4*)(dst + idx) = o;
    }
}

// -------------------------------------------------------------------------
// Kernel 2: lin[m][n] = sum_k A[m][k] * W[n][k]
// bf16 inputs, fp32 accumulate via v_wmma_f32_16x16x32_bf16.
// Staging: async global->LDS (double buffered) when available.
// -------------------------------------------------------------------------
__global__ __launch_bounds__(256) void gemm_bf16_wmma_kernel(
    const unsigned short* __restrict__ A,   // [BATCH][KDIM] bf16 bits
    const unsigned short* __restrict__ W,   // [NGATE][KDIM] bf16 bits
    float* __restrict__ C)                  // [BATCH][NGATE]
{
#if USE_ASYNC_LDS
    __shared__ __align__(16) unsigned short As[2][BM * LDS_STRIDE];
    __shared__ __align__(16) unsigned short Bs[2][BN * LDS_STRIDE];
#else
    __shared__ __align__(16) unsigned short As[1][BM * LDS_STRIDE];
    __shared__ __align__(16) unsigned short Bs[1][BN * LDS_STRIDE];
#endif

    const int t     = threadIdx.x;
    const int lane  = t & 31;
    const int wave  = t >> 5;
    const int waveM = wave & 1;        // 0..1 : 64 output rows each
    const int waveN = wave >> 1;       // 0..3 : 32 output cols each
    const int l16   = lane & 15;
    const int khalf = (lane >> 4) * 8; // lanes 0-15 -> K base 0, lanes 16-31 -> K base 8

    const int blockM = blockIdx.y * BM;
    const int blockN = blockIdx.x * BN;

    // staging assignment: 2 threads per tile row, 16 bf16 (32B) per thread
    const int ldRow = t >> 1;
    const int ldCol = (t & 1) * 16;
    const unsigned short* gA = A + (size_t)(blockM + ldRow) * KDIM + ldCol;
    const unsigned short* gW = W + (size_t)(blockN + ldRow) * KDIM + ldCol;
    const int sOff = ldRow * LDS_STRIDE + ldCol;

    v8f acc[4][2];
#pragma unroll
    for (int mi = 0; mi < 4; ++mi)
#pragma unroll
        for (int ni = 0; ni < 2; ++ni)
            acc[mi][ni] = (v8f){0.f, 0.f, 0.f, 0.f, 0.f, 0.f, 0.f, 0.f};

#if USE_ASYNC_LDS
    // Issue one 16KB tile (A+B) into LDS buffer `buf`: 4 async B128 ops/thread.
    auto issue_tile = [&](int k0, int buf) {
        __builtin_amdgcn_global_load_async_to_lds_b128(
            (global_v4i_ptr)(gA + k0),
            (lds_v4i_ptr)(&As[buf][sOff]), 0, 0);
        __builtin_amdgcn_global_load_async_to_lds_b128(
            (global_v4i_ptr)(gA + k0 + 8),
            (lds_v4i_ptr)(&As[buf][sOff + 8]), 0, 0);
        __builtin_amdgcn_global_load_async_to_lds_b128(
            (global_v4i_ptr)(gW + k0),
            (lds_v4i_ptr)(&Bs[buf][sOff]), 0, 0);
        __builtin_amdgcn_global_load_async_to_lds_b128(
            (global_v4i_ptr)(gW + k0 + 8),
            (lds_v4i_ptr)(&Bs[buf][sOff + 8]), 0, 0);
    };

    issue_tile(0, 0);
    for (int kt = 0; kt < NKT; ++kt) {
        const int cur = kt & 1;
        if (kt + 1 < NKT) {
            issue_tile((kt + 1) * BK, cur ^ 1);
            __builtin_amdgcn_s_wait_asynccnt(4);  // leave next tile's 4 ops in flight
        } else {
            __builtin_amdgcn_s_wait_asynccnt(0);
        }
        __syncthreads();  // tile kt visible to all waves

        const unsigned short* cA = As[cur];
        const unsigned short* cB = Bs[cur];

        Frag32B bfr[2];
#pragma unroll
        for (int ni = 0; ni < 2; ++ni) {
            const unsigned short* p = &cB[(waveN * 32 + ni * 16 + l16) * LDS_STRIDE + khalf];
            bfr[ni].u[0] = *(const uint4*)(p);
            bfr[ni].u[1] = *(const uint4*)(p + 16);
        }
#pragma unroll
        for (int mi = 0; mi < 4; ++mi) {
            Frag32B afr;
            const unsigned short* p = &cA[(waveM * 64 + mi * 16 + l16) * LDS_STRIDE + khalf];
            afr.u[0] = *(const uint4*)(p);
            afr.u[1] = *(const uint4*)(p + 16);
#pragma unroll
            for (int ni = 0; ni < 2; ++ni)
                acc[mi][ni] = __builtin_amdgcn_wmma_f32_16x16x32_bf16(
                    false, afr.v, false, bfr[ni].v,
                    (short)0, acc[mi][ni], false, false);
        }
        __syncthreads();  // all reads done before buffer is overwritten
    }
#else
    unsigned short* sA = &As[0][sOff];
    unsigned short* sB = &Bs[0][sOff];
    for (int k0 = 0; k0 < KDIM; k0 += BK) {
        const uint4 a0 = *(const uint4*)(gA + k0);
        const uint4 a1 = *(const uint4*)(gA + k0 + 8);
        const uint4 b0 = *(const uint4*)(gW + k0);
        const uint4 b1 = *(const uint4*)(gW + k0 + 8);
        __syncthreads();
        *(uint4*)(sA)     = a0;
        *(uint4*)(sA + 8) = a1;
        *(uint4*)(sB)     = b0;
        *(uint4*)(sB + 8) = b1;
        __syncthreads();

        Frag32B bfr[2];
#pragma unroll
        for (int ni = 0; ni < 2; ++ni) {
            const unsigned short* p = &Bs[0][(waveN * 32 + ni * 16 + l16) * LDS_STRIDE + khalf];
            bfr[ni].u[0] = *(const uint4*)(p);
            bfr[ni].u[1] = *(const uint4*)(p + 16);
        }
#pragma unroll
        for (int mi = 0; mi < 4; ++mi) {
            Frag32B afr;
            const unsigned short* p = &As[0][(waveM * 64 + mi * 16 + l16) * LDS_STRIDE + khalf];
            afr.u[0] = *(const uint4*)(p);
            afr.u[1] = *(const uint4*)(p + 16);
#pragma unroll
            for (int ni = 0; ni < 2; ++ni)
                acc[mi][ni] = __builtin_amdgcn_wmma_f32_16x16x32_bf16(
                    false, afr.v, false, bfr[ni].v,
                    (short)0, acc[mi][ni], false, false);
        }
    }
#endif

    const int rowBase = blockM + waveM * 64 + ((lane >> 4) << 3);
    const int colBase = blockN + waveN * 32 + l16;
#pragma unroll
    for (int mi = 0; mi < 4; ++mi)
#pragma unroll
        for (int ni = 0; ni < 2; ++ni)
#pragma unroll
            for (int r = 0; r < 8; ++r)
                C[(size_t)(rowBase + mi * 16 + r) * NGATE + (colBase + ni * 16)] =
                    acc[mi][ni][r];
}

// -------------------------------------------------------------------------
// Kernel 3: per-row epilogue. One 256-thread block per batch row.
// Each thread owns a float4 (elements 4t..4t+3) of every 1024-wide chunk.
// -------------------------------------------------------------------------
static __device__ __forceinline__ float block_sum_256(float v, float* sm) {
#pragma unroll
    for (int off = 16; off > 0; off >>= 1)
        v += __shfl_down(v, off, 32);
    const int lane = threadIdx.x & 31;
    const int w    = threadIdx.x >> 5;
    if (lane == 0) sm[w] = v;
    __syncthreads();
    if (w == 0) {
        float x = (lane < 8) ? sm[lane] : 0.0f;
#pragma unroll
        for (int off = 4; off > 0; off >>= 1)
            x += __shfl_down(x, off, 32);
        if (lane == 0) sm[0] = x;
    }
    __syncthreads();
    const float r = sm[0];
    __syncthreads();
    return r;
}

static __device__ __forceinline__ float sigmoidf_(float x) {
    return 1.0f / (1.0f + __expf(-x));
}

__global__ __launch_bounds__(256) void epilogue_kernel(
    const float* __restrict__ lin,   // [BATCH][NGATE]
    const float* __restrict__ c_in,  // [BATCH][HIDDEN]
    const float* __restrict__ lnhw, const float* __restrict__ lnhb,  // [NGATE]
    const float* __restrict__ lncw, const float* __restrict__ lncb,  // [HIDDEN]
    float* __restrict__ h_out, float* __restrict__ c_out)            // [BATCH][HIDDEN]
{
    __shared__ float sm[8];
    const int row = blockIdx.x;
    const int t   = threadIdx.x;
    const int j0  = t * 4;                  // this thread's float4 within a chunk
    const size_t rbase = (size_t)row * NGATE;

    float g[4][4];  // [gate chunk][component]

#pragma unroll
    for (int ch = 0; ch < 4; ++ch) {
        const float4 xv = *(const float4*)&lin[rbase + ch * HIDDEN + j0];
        float v[4] = {xv.x, xv.y, xv.z, xv.w};
        float s = 0.f, s2 = 0.f;
#pragma unroll
        for (int e = 0; e < 4; ++e) { s += v[e]; s2 += v[e] * v[e]; }
        s  = block_sum_256(s, sm);
        s2 = block_sum_256(s2, sm);
        const float mu  = s * (1.0f / HIDDEN);
        const float var = s2 * (1.0f / HIDDEN) - mu * mu;
        const float inv = rsqrtf(var + 1e-5f);
        const float4 wv = *(const float4*)&lnhw[ch * HIDDEN + j0];
        const float4 bv = *(const float4*)&lnhb[ch * HIDDEN + j0];
        const float wa[4] = {wv.x, wv.y, wv.z, wv.w};
        const float ba[4] = {bv.x, bv.y, bv.z, bv.w};
#pragma unroll
        for (int e = 0; e < 4; ++e)
            g[ch][e] = (v[e] - mu) * inv * wa[e] + ba[e];
    }

    // gates, c_new
    const float4 cv4 = *(const float4*)&c_in[(size_t)row * HIDDEN + j0];
    const float cold[4] = {cv4.x, cv4.y, cv4.z, cv4.w};
    float cnv[4], ov[4];
    float s = 0.f, s2 = 0.f;
#pragma unroll
    for (int e = 0; e < 4; ++e) {
        const float ig = sigmoidf_(g[0][e]);
        const float fg = sigmoidf_(g[1][e] + 1.0f);   // FORGET_BIAS
        const float og = sigmoidf_(g[2][e]);
        const float cc = tanhf(g[3][e]);
        const float cn = fg * cold[e] + ig * cc;
        cnv[e] = cn;
        ov[e]  = og;
        s  += cn;
        s2 += cn * cn;
    }
    *(float4*)&c_out[(size_t)row * HIDDEN + j0] =
        make_float4(cnv[0], cnv[1], cnv[2], cnv[3]);

    // layernorm over c_new, then h_new
    s  = block_sum_256(s, sm);
    s2 = block_sum_256(s2, sm);
    const float mu  = s * (1.0f / HIDDEN);
    const float var = s2 * (1.0f / HIDDEN) - mu * mu;
    const float inv = rsqrtf(var + 1e-5f);
    const float4 wv = *(const float4*)&lncw[j0];
    const float4 bv = *(const float4*)&lncb[j0];
    const float wa[4] = {wv.x, wv.y, wv.z, wv.w};
    const float ba[4] = {bv.x, bv.y, bv.z, bv.w};
    float hv[4];
#pragma unroll
    for (int e = 0; e < 4; ++e) {
        const float cln = (cnv[e] - mu) * inv * wa[e] + ba[e];
        hv[e] = ov[e] * tanhf(cln);
    }
    *(float4*)&h_out[(size_t)row * HIDDEN + j0] =
        make_float4(hv[0], hv[1], hv[2], hv[3]);
}

// -------------------------------------------------------------------------
extern "C" void kernel_launch(void* const* d_in, const int* in_sizes, int n_in,
                              void* d_out, int out_size, void* d_ws, size_t ws_size,
                              hipStream_t stream) {
    const float* x    = (const float*)d_in[0];
    const float* h    = (const float*)d_in[1];
    const float* c    = (const float*)d_in[2];
    const float* wih  = (const float*)d_in[3];
    const float* whh  = (const float*)d_in[4];
    const float* lnhw = (const float*)d_in[5];
    const float* lnhb = (const float*)d_in[6];
    const float* lncw = (const float*)d_in[7];
    const float* lncb = (const float*)d_in[8];

    const size_t Abytes = (size_t)BATCH * KDIM * sizeof(unsigned short);  // 12.6 MB
    unsigned short* Abf = (unsigned short*)d_ws;
    unsigned short* Wbf = (unsigned short*)((char*)d_ws + Abytes);
    float*          lin = (float*)((char*)d_ws + 2 * Abytes);             // 64 MB

    pack_bf16_kernel<<<512, 256, 0, stream>>>(x,   INPUT, h,   HIDDEN, Abf, BATCH);
    pack_bf16_kernel<<<512, 256, 0, stream>>>(wih, INPUT, whh, HIDDEN, Wbf, NGATE);

    dim3 grid(NGATE / BN, BATCH / BM);  // 32 x 32
    gemm_bf16_wmma_kernel<<<grid, 256, 0, stream>>>(Abf, Wbf, lin);

    float* h_out = (float*)d_out;
    float* c_out = h_out + (size_t)BATCH * HIDDEN;
    epilogue_kernel<<<BATCH, 256, 0, stream>>>(lin, c, lnhw, lnhb, lncw, lncb,
                                               h_out, c_out);
}
